// CausalSelfAttention_28853590295171
// MI455X (gfx1250) — matrix operand order repaired and checked by reading, not compiled, not run on hardware
//
#include <hip/hip_runtime.h>
#include <math.h>

// ---------------------------------------------------------------------------
// Causal self-attention for MI455X (gfx1250, wave32, WMMA + TDM).
//   B=4, T=2048, E=512, H=8, D=64.  ~52 GFLOP -> matrix-core bound; all GEMMs
//   and attention einsums use v_wmma_f32_16x16x32_f16 (f16 in, f32 acc).
//   Flash-attention K/V tiles are DMA'd into LDS by the Tensor Data Mover
//   (tensor_load_to_lds, TENSORcnt) with double buffering.
// ---------------------------------------------------------------------------

typedef __attribute__((ext_vector_type(16))) _Float16 v16h;
typedef __attribute__((ext_vector_type(2)))  _Float16 h2;
typedef __attribute__((ext_vector_type(8)))  float    v8f;
typedef unsigned int u32x4 __attribute__((ext_vector_type(4)));
typedef int          i32x4 __attribute__((ext_vector_type(4)));
typedef int          i32x8 __attribute__((ext_vector_type(8)));

#define WMMA_F32_F16(a, b, c) \
    __builtin_amdgcn_wmma_f32_16x16x32_f16(false, (a), false, (b), (short)0, (c), false, false)

#if defined(__has_builtin)
#if __has_builtin(__builtin_amdgcn_tensor_load_to_lds)
#define HAVE_TDM 1
#endif
#endif

constexpr int Bb = 4, T = 2048, E = 512, H = 8, D = 64, E3 = 1536;
constexpr int M_TOTAL = Bb * T; // 8192

#ifdef HAVE_TDM
// Issue a TDM 2-D tile load: rows x cols f16 elements, row stride in elems.
// D# layout per CDNA5 ISA ch.8 (group0: count/lds/global/type, group1: dims).
__device__ __forceinline__ void tdm_load_2d_f16(unsigned int lds_byte_addr,
                                                const _Float16* gptr,
                                                int rows, int cols,
                                                int row_stride_elems) {
    unsigned long long ga = (unsigned long long)(uintptr_t)gptr;
    u32x4 g0 = {};
    g0[0] = 1u;                                    // count=1, user mode, no gather
    g0[1] = lds_byte_addr;                         // [63:32]  lds_addr
    g0[2] = (unsigned int)ga;                      // [95:64]  global_addr lo
    g0[3] = ((unsigned int)(ga >> 32) & 0x01FFFFFFu) | (2u << 30); // addr hi | type=2
    i32x8 g1 = {};
    g1[0] = 1 << 16;                               // data_size=1 (2 bytes)
    g1[1] = (cols & 0xFFFF) << 16;                 // tensor_dim0[15:0] @ [63:48]
    g1[2] = ((cols >> 16) & 0xFFFF) | ((rows & 0xFFFF) << 16); // dim0 hi | dim1 lo
    g1[3] = ((rows >> 16) & 0xFFFF) | ((cols & 0xFFFF) << 16); // dim1 hi | tile_dim0
    g1[4] = rows & 0xFFFF;                         // tile_dim1 (tile_dim2 = 0)
    g1[5] = row_stride_elems;                      // tensor_dim0_stride lo32
    i32x4 gz = {};
#if __clang_major__ >= 23
    i32x8 gz8 = {};
    __builtin_amdgcn_tensor_load_to_lds(g0, g1, gz, gz, gz8, 0);
#else
    __builtin_amdgcn_tensor_load_to_lds(g0, g1, gz, gz, 0);
#endif
}
#endif

// ---------------------------------------------------------------------------
// Kernel 1: qkv = x @ c_attn_w + b   (M=8192, N=1536, K=512)
// Block tile 128(M) x 32(N), K-step 32.  8 waves, each owns a 16-row stripe
// and two 16x16 N tiles.  Output scattered to Q/K/V f16 [B,H,T,D] in ws.
// ---------------------------------------------------------------------------
__global__ __launch_bounds__(256) void qkv_gemm(const float* __restrict__ x,
                                                const float* __restrict__ w,
                                                const float* __restrict__ bias,
                                                _Float16* __restrict__ Qo,
                                                _Float16* __restrict__ Ko,
                                                _Float16* __restrict__ Vo) {
    __shared__ _Float16 As[128 * 32];   // x tile,  row-major [m][k]
    __shared__ _Float16 Bs[32 * 32];    // w tile,  transposed [n][k]

    const int tid   = threadIdx.x;
    const int wave  = tid >> 5, lane = tid & 31;
    const int lhalf = lane >> 4, l16 = lane & 15;
    const int nbase = blockIdx.x * 32;
    const int mbase = blockIdx.y * 128;

    v8f acc0 = {}, acc1 = {};

    for (int ks = 0; ks < E; ks += 32) {
        #pragma unroll
        for (int t = 0; t < 16; ++t) {                // 128x32 A elems / 256 thr
            int e = tid + t * 256;
            int r = e >> 5, c = e & 31;
            As[e] = (_Float16)x[(size_t)(mbase + r) * E + ks + c];
        }
        #pragma unroll
        for (int t = 0; t < 4; ++t) {                 // 32x32 B elems / 256 thr
            int e = tid + t * 256;
            int k = e >> 5, n = e & 31;
            Bs[n * 32 + k] = (_Float16)w[(size_t)(ks + k) * E3 + nbase + n];
        }
        __syncthreads();

        // A fragment: lane holds row (wave*16 + l16); ISA 16-bit A swizzle.
        v16h af;
        #pragma unroll
        for (int v = 0; v < 8; ++v) {
            int kd = (v < 4 ? 0 : 16) + lhalf * 8 + (v & 3) * 2;
            h2 p = *(const h2*)&As[(wave * 16 + l16) * 32 + kd];
            af[2 * v] = p.x; af[2 * v + 1] = p.y;
        }
        #pragma unroll
        for (int nt = 0; nt < 2; ++nt) {
            v16h bf;   // B fragment: lane holds col (nt*16 + l16)
            #pragma unroll
            for (int v = 0; v < 8; ++v) {
                int kd = (v < 4 ? 0 : 16) + lhalf * 8 + (v & 3) * 2;
                h2 p = *(const h2*)&Bs[(nt * 16 + l16) * 32 + kd];
                bf[2 * v] = p.x; bf[2 * v + 1] = p.y;
            }
            if (nt == 0) acc0 = WMMA_F32_F16(af, bf, acc0);
            else         acc1 = WMMA_F32_F16(af, bf, acc1);
        }
        __syncthreads();
    }

    // Epilogue: bias add, scatter to [B,H,T,D] f16 (q / k / v split).
    #pragma unroll
    for (int nt = 0; nt < 2; ++nt) {
        const v8f& acc = (nt == 0) ? acc0 : acc1;
        int j = nbase + nt * 16 + l16;               // C-layout col
        float bj = bias[j];
        int which = j / E;
        int e = j % E;
        int h = e / D, d = e % D;
        _Float16* dst = (which == 0) ? Qo : (which == 1 ? Ko : Vo);
        #pragma unroll
        for (int v = 0; v < 8; ++v) {
            int m = mbase + wave * 16 + v + 8 * lhalf; // C-layout row
            int b = m / T, t = m % T;
            dst[((size_t)(b * H + h) * T + t) * D + d] = (_Float16)(acc[v] + bj);
        }
    }
}

// ---------------------------------------------------------------------------
// Kernel 2: flash attention.  One block per (b,h,128-query tile); 8 waves x
// 16 query rows.  K/V streamed 32 keys at a time into double-buffered LDS by
// the Tensor Data Mover (prefetch overlaps WMMA compute); online softmax.
// ---------------------------------------------------------------------------
__global__ __launch_bounds__(256) void flash_attn(const _Float16* __restrict__ Q,
                                                  const _Float16* __restrict__ K,
                                                  const _Float16* __restrict__ V,
                                                  _Float16* __restrict__ Y) {
    __shared__ _Float16 Kt[2][32 * 64];       // 32 keys x 64 d, double buffered
    __shared__ _Float16 Vt[2][32 * 64];
    __shared__ _Float16 Pt[8 * 16 * 32];      // per-wave P transpose scratch

    const int tid   = threadIdx.x;
    const int wave  = tid >> 5, lane = tid & 31;
    const int lhalf = lane >> 4, l16 = lane & 15;
    const int qt = blockIdx.x & 15;
    const int bh = blockIdx.x >> 4;
    const int qbase = qt * 128;
    const int qrow0 = qbase + wave * 16;
    const float scale = 0.125f;               // 1/sqrt(64)

    const _Float16* Kbh = K + (size_t)bh * T * D;
    const _Float16* Vbh = V + (size_t)bh * T * D;

    // Q fragments for the two 32-wide chunks of D (kept in VGPRs).
    const _Float16* qrow = Q + ((size_t)bh * T + qrow0 + l16) * D;
    v16h qa0, qa1;
    #pragma unroll
    for (int v = 0; v < 8; ++v) {
        int kd = (v < 4 ? 0 : 16) + lhalf * 8 + (v & 3) * 2;
        h2 p0 = *(const h2*)(qrow + kd);
        h2 p1 = *(const h2*)(qrow + 32 + kd);
        qa0[2 * v] = p0.x; qa0[2 * v + 1] = p0.y;
        qa1[2 * v] = p1.x; qa1[2 * v + 1] = p1.y;
    }

    v8f o0 = {}, o1 = {}, o2 = {}, o3 = {};
    float mi[8], li[8];
    #pragma unroll
    for (int v = 0; v < 8; ++v) { mi[v] = -3.0e38f; li[v] = 0.0f; }

    const int niter = (qbase + 128) / 32;     // block-uniform causal bound

#ifdef HAVE_TDM
    if (wave == 0) {                          // prologue: DMA tile 0 -> buf 0
        tdm_load_2d_f16((unsigned int)(uintptr_t)&Kt[0][0], Kbh, 32, 64, 64);
        tdm_load_2d_f16((unsigned int)(uintptr_t)&Vt[0][0], Vbh, 32, 64, 64);
    }
#endif

    for (int it = 0; it < niter; ++it) {
        const int kk  = it * 32;
        const int buf = it & 1;
        __syncthreads();   // everyone done reading buf^1 (iter it-1)

#ifdef HAVE_TDM
        if (wave == 0) {
            if (it + 1 < niter) {   // prefetch next tile into the other buffer
                tdm_load_2d_f16((unsigned int)(uintptr_t)&Kt[buf ^ 1][0],
                                Kbh + (size_t)(kk + 32) * D, 32, 64, 64);
                tdm_load_2d_f16((unsigned int)(uintptr_t)&Vt[buf ^ 1][0],
                                Vbh + (size_t)(kk + 32) * D, 32, 64, 64);
                // 4 outstanding, in-order: <=2 left means current pair landed
                __builtin_amdgcn_s_wait_tensorcnt((short)2);
            } else {
                __builtin_amdgcn_s_wait_tensorcnt((short)0);
            }
        }
#else
        #pragma unroll
        for (int t = 0; t < 8; ++t) {          // fallback cooperative staging
            int e = tid + t * 256;
            int kr = e >> 6, d = e & 63;
            Kt[buf][e] = Kbh[(size_t)(kk + kr) * D + d];
            Vt[buf][e] = Vbh[(size_t)(kk + kr) * D + d];
        }
#endif
        __syncthreads();   // publish tile `it` to all waves

        // S = Q . K^T  (two 16x16 key tiles, contraction over D=64 = 2 chains)
        v8f s0 = {}, s1 = {};
        #pragma unroll
        for (int c = 0; c < 2; ++c) {
            const v16h& qa = (c == 0) ? qa0 : qa1;
            v16h kb0, kb1;
            #pragma unroll
            for (int v = 0; v < 8; ++v) {
                int kd = c * 32 + (v < 4 ? 0 : 16) + lhalf * 8 + (v & 3) * 2;     // contiguous d pair
                h2 p0 = *(const h2*)&Kt[buf][l16 * 64 + kd];
                h2 p1 = *(const h2*)&Kt[buf][(16 + l16) * 64 + kd];
                kb0[2 * v] = p0.x; kb0[2 * v + 1] = p0.y;
                kb1[2 * v] = p1.x; kb1[2 * v + 1] = p1.y;
            }
            s0 = WMMA_F32_F16(qa, kb0, s0);
            s1 = WMMA_F32_F16(qa, kb1, s1);
        }

        // Online softmax; C-layout row for accumulator index v is v+8*lhalf,
        // striped across 16 lanes -> width-16 xor reductions.
        #pragma unroll
        for (int v = 0; v < 8; ++v) {
            int qg  = qrow0 + v + 8 * lhalf;
            float a0 = (kk + l16      <= qg) ? s0[v] * scale : -3.0e38f;
            float a1 = (kk + 16 + l16 <= qg) ? s1[v] * scale : -3.0e38f;
            float mx = fmaxf(a0, a1);
            #pragma unroll
            for (int wd = 1; wd < 16; wd <<= 1)
                mx = fmaxf(mx, __shfl_xor(mx, wd, 16));
            float newm  = fmaxf(mi[v], mx);
            float alpha = __expf(mi[v] - newm);
            float p0 = __expf(a0 - newm);
            float p1 = __expf(a1 - newm);
            float rs = p0 + p1;
            #pragma unroll
            for (int wd = 1; wd < 16; wd <<= 1)
                rs += __shfl_xor(rs, wd, 16);
            mi[v] = newm;
            li[v] = li[v] * alpha + rs;
            o0[v] *= alpha; o1[v] *= alpha; o2[v] *= alpha; o3[v] *= alpha;
            int r = v + 8 * lhalf;            // write P row-major in LDS
            Pt[wave * 512 + r * 32 + l16]      = (_Float16)p0;
            Pt[wave * 512 + r * 32 + 16 + l16] = (_Float16)p1;
        }

        // Re-read P with A-matrix swizzle (per-wave region, DS in-order).
        v16h pf;
        #pragma unroll
        for (int v = 0; v < 8; ++v) {
            int kd = (v < 4 ? 0 : 16) + lhalf * 8 + (v & 3) * 2;
            h2 p = *(const h2*)&Pt[wave * 512 + l16 * 32 + kd];
            pf[2 * v] = p.x; pf[2 * v + 1] = p.y;
        }

        // O += P . V  (four 16-wide d tiles)
        #pragma unroll
        for (int dt = 0; dt < 4; ++dt) {
            v16h vb;
            int d = dt * 16 + l16;
            #pragma unroll
            for (int u = 0; u < 8; ++u) {
                int k0 = (u < 4 ? 0 : 16) + lhalf * 8 + (u & 3) * 2;
                vb[2 * u]     = Vt[buf][k0 * 64 + d];
                vb[2 * u + 1] = Vt[buf][(k0 + 1) * 64 + d];
            }
            v8f& o = (dt == 0) ? o0 : (dt == 1) ? o1 : (dt == 2) ? o2 : o3;
            o = WMMA_F32_F16(pf, vb, o);
        }
    }

    // Epilogue: normalize and store Y f16 as [B,T,E] with e = h*64 + d.
    const int b = bh >> 3, h = bh & 7;
    #pragma unroll
    for (int dt = 0; dt < 4; ++dt) {
        const v8f& o = (dt == 0) ? o0 : (dt == 1) ? o1 : (dt == 2) ? o2 : o3;
        #pragma unroll
        for (int v = 0; v < 8; ++v) {
            int qg = qrow0 + v + 8 * lhalf;
            Y[((size_t)b * T + qg) * E + h * D + dt * 16 + l16] =
                (_Float16)(o[v] / li[v]);
        }
    }
}

// ---------------------------------------------------------------------------
// Kernel 3: out = Y @ c_proj_w + b   (M=8192, N=512, K=512), f32 output.
// ---------------------------------------------------------------------------
__global__ __launch_bounds__(256) void proj_gemm(const _Float16* __restrict__ Y,
                                                 const float* __restrict__ w,
                                                 const float* __restrict__ bias,
                                                 float* __restrict__ out) {
    __shared__ _Float16 As[128 * 32];
    __shared__ _Float16 Bs[32 * 32];

    const int tid   = threadIdx.x;
    const int wave  = tid >> 5, lane = tid & 31;
    const int lhalf = lane >> 4, l16 = lane & 15;
    const int nbase = blockIdx.x * 32;
    const int mbase = blockIdx.y * 128;

    v8f acc0 = {}, acc1 = {};

    for (int ks = 0; ks < E; ks += 32) {
        #pragma unroll
        for (int t = 0; t < 16; ++t) {
            int e = tid + t * 256;
            int r = e >> 5, c = e & 31;
            As[e] = Y[(size_t)(mbase + r) * E + ks + c];
        }
        #pragma unroll
        for (int t = 0; t < 4; ++t) {
            int e = tid + t * 256;
            int k = e >> 5, n = e & 31;
            Bs[n * 32 + k] = (_Float16)w[(size_t)(ks + k) * E + nbase + n];
        }
        __syncthreads();

        v16h af;
        #pragma unroll
        for (int v = 0; v < 8; ++v) {
            int kd = (v < 4 ? 0 : 16) + lhalf * 8 + (v & 3) * 2;
            h2 p = *(const h2*)&As[(wave * 16 + l16) * 32 + kd];
            af[2 * v] = p.x; af[2 * v + 1] = p.y;
        }
        #pragma unroll
        for (int nt = 0; nt < 2; ++nt) {
            v16h bf;
            #pragma unroll
            for (int v = 0; v < 8; ++v) {
                int kd = (v < 4 ? 0 : 16) + lhalf * 8 + (v & 3) * 2;
                h2 p = *(const h2*)&Bs[(nt * 16 + l16) * 32 + kd];
                bf[2 * v] = p.x; bf[2 * v + 1] = p.y;
            }
            if (nt == 0) acc0 = WMMA_F32_F16(af, bf, acc0);
            else         acc1 = WMMA_F32_F16(af, bf, acc1);
        }
        __syncthreads();
    }

    #pragma unroll
    for (int nt = 0; nt < 2; ++nt) {
        const v8f& acc = (nt == 0) ? acc0 : acc1;
        int j = nbase + nt * 16 + l16;
        float bj = bias[j];
        #pragma unroll
        for (int v = 0; v < 8; ++v) {
            int m = mbase + wave * 16 + v + 8 * lhalf;
            out[(size_t)m * E + j] = acc[v] + bj;
        }
    }
}

// ---------------------------------------------------------------------------
extern "C" void kernel_launch(void* const* d_in, const int* in_sizes, int n_in,
                              void* d_out, int out_size, void* d_ws, size_t ws_size,
                              hipStream_t stream) {
    (void)in_sizes; (void)n_in; (void)out_size; (void)ws_size;
    const float* x       = (const float*)d_in[0];
    const float* attn_w  = (const float*)d_in[1];
    const float* attn_b  = (const float*)d_in[2];
    const float* proj_w  = (const float*)d_in[3];
    const float* proj_b  = (const float*)d_in[4];
    float*       out     = (float*)d_out;

    const size_t perMat = (size_t)Bb * H * T * D;     // 4 Mi halfs = 8 MB
    _Float16* Q = (_Float16*)d_ws;
    _Float16* K = Q + perMat;
    _Float16* V = K + perMat;
    _Float16* Y = V + perMat;                          // total 32 MB of ws

    qkv_gemm<<<dim3(E3 / 32, M_TOTAL / 128), 256, 0, stream>>>(x, attn_w, attn_b, Q, K, V);
    flash_attn<<<dim3((Bb * H) * (T / 128)), 256, 0, stream>>>(Q, K, V, Y);
    proj_gemm<<<dim3(E / 32, M_TOTAL / 128), 256, 0, stream>>>(Y, proj_w, proj_b, out);
}
